// EncoderLayer_31275951849563
// MI455X (gfx1250) — matrix operand order repaired and checked by reading, not compiled, not run on hardware
//
#include <hip/hip_runtime.h>

// ---------------------------------------------------------------------------
// Types for CDNA5 WMMA (wave32, 16x16x32 bf16 -> f32)
// ---------------------------------------------------------------------------
typedef unsigned short bfraw;
typedef __attribute__((ext_vector_type(16))) __bf16 v16bf;
typedef __attribute__((ext_vector_type(8)))  __bf16 v8bf;
typedef __attribute__((ext_vector_type(8)))  float  v8f;
typedef __attribute__((ext_vector_type(4)))  int    v4i_a;

union BF16x16 { v16bf v; v8bf h[2]; };

__device__ __forceinline__ bfraw f2bf(float f) {
  unsigned u = __float_as_uint(f);
  u += 0x7FFFu + ((u >> 16) & 1u);        // round-to-nearest-even
  return (bfraw)(u >> 16);
}

#ifndef __has_builtin
#define __has_builtin(x) 0
#endif
#if __has_builtin(__builtin_amdgcn_global_load_async_to_lds_b128)
#define HAVE_ASYNC_LDS 1
#else
#define HAVE_ASYNC_LDS 0
#endif

#if HAVE_ASYNC_LDS
__device__ __forceinline__ void wait_async0() {
#if __has_builtin(__builtin_amdgcn_s_wait_asynccnt)
  __builtin_amdgcn_s_wait_asynccnt(0);
#else
  asm volatile("s_wait_asynccnt 0x0" ::: "memory");
#endif
}
typedef __attribute__((address_space(1))) v4i_a* gv4i_t;   // global int4*
typedef __attribute__((address_space(3))) v4i_a* lv4i_t;   // LDS int4*
__device__ __forceinline__ void async_copy16(const bfraw* g, bfraw* l) {
  bfraw* gnc = const_cast<bfraw*>(g);
  __builtin_amdgcn_global_load_async_to_lds_b128((gv4i_t)gnc, (lv4i_t)l, 0, 0);
}
#endif

// Problem constants (from reference)
constexpr int       Bn   = 8;
constexpr int       Sn   = 4096;
constexpr int       Dn   = 512;
constexpr int       Hn   = 8;
constexpr int       DKn  = 64;
constexpr int       TOPK = 8;                       // int(log(4096)) == 8
constexpr long long Mrows = (long long)Bn * Sn;     // 32768
constexpr long long BSD   = Mrows * Dn;             // 16,777,216

// ---------------------------------------------------------------------------
// f32 -> bf16 cast
// ---------------------------------------------------------------------------
__global__ __launch_bounds__(256) void cast_f32_bf16(
    const float* __restrict__ in, bfraw* __restrict__ outp, long long n) {
  long long i = (long long)blockIdx.x * 256 + threadIdx.x;
  if (i < n) outp[i] = f2bf(in[i]);
}

// ---------------------------------------------------------------------------
// Tiled bf16 WMMA GEMM:  C(MxN) = A(MxK) * B(KxN) + bias [+ resid] [relu]
// 128x128 tile per workgroup, 8 waves, each 32x64 (2x4 WMMA tiles), BK=64
// (two K=32 WMMA steps per stage). LDS double-buffered; A tile staged with
// GLOBAL_LOAD_ASYNC_TO_LDS_B128 when available (ASYNCcnt), B staged through
// registers (needs transpose to [N][K]).
// ---------------------------------------------------------------------------
#define LDT 72  // LDS row stride in bf16 (64 + 8 pad; 144B keeps 16B align)

template<bool RELU, bool RESID, bool F32OUT, bool BF16OUT>
__global__ __launch_bounds__(256) void gemm_bf16_wmma(
    const bfraw* __restrict__ A, const bfraw* __restrict__ Bw,
    const float* __restrict__ bias, const float* __restrict__ resid,
    float* __restrict__ Cf, bfraw* __restrict__ Cb, int N, int K) {
  __shared__ __align__(16) bfraw As[2][128 * LDT];   // [M][K]
  __shared__ __align__(16) bfraw Bs[2][128 * LDT];   // [N][K] (transposed)

  const int tid   = threadIdx.x;
  const int wave  = tid >> 5;
  const int lane  = tid & 31;
  const int waveM = (wave >> 1) * 32;
  const int waveN = (wave & 1) * 64;
  const int bm    = blockIdx.y * 128;
  const int bn    = blockIdx.x * 128;

  v8f acc[2][4];
  #pragma unroll
  for (int nt = 0; nt < 4; ++nt) {
    float bvv = bias[bn + waveN + nt * 16 + (lane & 15)];
    #pragma unroll
    for (int mt = 0; mt < 2; ++mt)
      #pragma unroll
      for (int r = 0; r < 8; ++r) acc[mt][nt][r] = bvv;
  }

  // -------- prologue: stage tile 0 into buffer 0
  {
#if HAVE_ASYNC_LDS
    #pragma unroll
    for (int it = 0; it < 4; ++it) {
      int q = tid + it * 256;
      int row = q >> 3, kc = (q & 7) * 8;
      async_copy16(&A[(size_t)(bm + row) * K + kc], &As[0][row * LDT + kc]);
    }
#else
    uint4 aR[4];
    #pragma unroll
    for (int it = 0; it < 4; ++it) {
      int q = tid + it * 256;
      int row = q >> 3, kc = (q & 7) * 8;
      aR[it] = *(const uint4*)&A[(size_t)(bm + row) * K + kc];
    }
#endif
    uint4 bR[4];
    #pragma unroll
    for (int it = 0; it < 4; ++it) {
      int q = tid + it * 256;
      int kr = q >> 4, nq = (q & 15) * 8;
      bR[it] = *(const uint4*)&Bw[(size_t)kr * N + bn + nq];
    }
#if !HAVE_ASYNC_LDS
    #pragma unroll
    for (int it = 0; it < 4; ++it) {
      int q = tid + it * 256;
      int row = q >> 3, kc = (q & 7) * 8;
      *(uint4*)&As[0][row * LDT + kc] = aR[it];
    }
#endif
    #pragma unroll
    for (int it = 0; it < 4; ++it) {
      int q = tid + it * 256;
      int kr = q >> 4, nq = (q & 15) * 8;
      unsigned w[4] = {bR[it].x, bR[it].y, bR[it].z, bR[it].w};
      #pragma unroll
      for (int j = 0; j < 4; ++j) {
        Bs[0][(nq + 2 * j)     * LDT + kr] = (bfraw)(w[j] & 0xFFFFu);
        Bs[0][(nq + 2 * j + 1) * LDT + kr] = (bfraw)(w[j] >> 16);
      }
    }
#if HAVE_ASYNC_LDS
    wait_async0();
#endif
  }
  __syncthreads();

  int buf = 0;
  for (int k0 = 0; k0 < K; k0 += 64) {
    const bool more = (k0 + 64) < K;
    const int  nxt  = buf ^ 1;
    const int  k1   = k0 + 64;

    uint4 bR[4];
#if !HAVE_ASYNC_LDS
    uint4 aR[4];
#endif
    if (more) {
#if HAVE_ASYNC_LDS
      #pragma unroll
      for (int it = 0; it < 4; ++it) {
        int q = tid + it * 256;
        int row = q >> 3, kc = (q & 7) * 8;
        async_copy16(&A[(size_t)(bm + row) * K + k1 + kc],
                     &As[nxt][row * LDT + kc]);
      }
#else
      #pragma unroll
      for (int it = 0; it < 4; ++it) {
        int q = tid + it * 256;
        int row = q >> 3, kc = (q & 7) * 8;
        aR[it] = *(const uint4*)&A[(size_t)(bm + row) * K + k1 + kc];
      }
#endif
      #pragma unroll
      for (int it = 0; it < 4; ++it) {
        int q = tid + it * 256;
        int kr = q >> 4, nq = (q & 15) * 8;
        bR[it] = *(const uint4*)&Bw[(size_t)(k1 + kr) * N + bn + nq];
      }
    }

    // -------- compute: two K=32 WMMA steps from buffer `buf`
    #pragma unroll
    for (int kk = 0; kk < 64; kk += 32) {
      const int mIdx  = lane & 15;
      const int kHalf = (lane >> 4) * 8;   // A: lanes 0-15 K{0..7,16..23}
      const int kBase = (lane >> 4) * 16;  // B: lanes 0-15 K{0..7,16..23}
      BF16x16 aF[2], bF[4];
      #pragma unroll
      for (int mt = 0; mt < 2; ++mt) {
        int m = waveM + mt * 16 + mIdx;
        aF[mt].h[0] = *(const v8bf*)&As[buf][m * LDT + kk + kHalf];
        aF[mt].h[1] = *(const v8bf*)&As[buf][m * LDT + kk + 16 + kHalf];
      }
      #pragma unroll
      for (int nt = 0; nt < 4; ++nt) {
        int n = waveN + nt * 16 + mIdx;
        bF[nt].h[0] = *(const v8bf*)&Bs[buf][n * LDT + kk + kHalf];
        bF[nt].h[1] = *(const v8bf*)&Bs[buf][n * LDT + kk + 16 + kHalf];
      }
      #pragma unroll
      for (int mt = 0; mt < 2; ++mt)
        #pragma unroll
        for (int nt = 0; nt < 4; ++nt)
          acc[mt][nt] = __builtin_amdgcn_wmma_f32_16x16x32_bf16(
              false, aF[mt].v, false, bF[nt].v, (short)0, acc[mt][nt],
              false, false);
    }

    // -------- commit next tile to LDS
    if (more) {
#if !HAVE_ASYNC_LDS
      #pragma unroll
      for (int it = 0; it < 4; ++it) {
        int q = tid + it * 256;
        int row = q >> 3, kc = (q & 7) * 8;
        *(uint4*)&As[nxt][row * LDT + kc] = aR[it];
      }
#endif
      #pragma unroll
      for (int it = 0; it < 4; ++it) {
        int q = tid + it * 256;
        int kr = q >> 4, nq = (q & 15) * 8;
        unsigned w[4] = {bR[it].x, bR[it].y, bR[it].z, bR[it].w};
        #pragma unroll
        for (int j = 0; j < 4; ++j) {
          Bs[nxt][(nq + 2 * j)     * LDT + kr] = (bfraw)(w[j] & 0xFFFFu);
          Bs[nxt][(nq + 2 * j + 1) * LDT + kr] = (bfraw)(w[j] >> 16);
        }
      }
#if HAVE_ASYNC_LDS
      wait_async0();
#endif
    }
    __syncthreads();
    buf = nxt;
  }

  // -------- epilogue: C layout = lane -> N, VGPR r -> M row
  const int nIdx = lane & 15;
  const int mOff = (lane >> 4) * 8;
  #pragma unroll
  for (int mt = 0; mt < 2; ++mt) {
    #pragma unroll
    for (int nt = 0; nt < 4; ++nt) {
      int gn = bn + waveN + nt * 16 + nIdx;
      #pragma unroll
      for (int r = 0; r < 8; ++r) {
        int gm = bm + waveM + mt * 16 + mOff + r;
        float v = acc[mt][nt][r];
        if (RESID) v += resid[(size_t)gm * N + gn];
        if (RELU)  v = v > 0.f ? v : 0.f;
        if (F32OUT)  Cf[(size_t)gm * N + gn] = v;
        if (BF16OUT) Cb[(size_t)gm * N + gn] = f2bf(v);
      }
    }
  }
}

// ---------------------------------------------------------------------------
// (B,S,D) -> (B,D,S) fp32 transpose (32x32 LDS tiles)
// ---------------------------------------------------------------------------
__global__ void transpose_sd_kernel(const float* __restrict__ in,
                                    float* __restrict__ outp) {
  __shared__ float tile[32][33];
  const int b  = blockIdx.z;
  const int s0 = blockIdx.x * 32, d0 = blockIdx.y * 32;
  const int tx = threadIdx.x, ty = threadIdx.y;
  #pragma unroll
  for (int j = 0; j < 32; j += 8)
    tile[ty + j][tx] = in[((long long)b * Sn + (s0 + ty + j)) * Dn + d0 + tx];
  __syncthreads();
  #pragma unroll
  for (int j = 0; j < 32; j += 8)
    outp[((long long)b * Dn + (d0 + ty + j)) * Sn + s0 + tx] = tile[tx][ty + j];
}

// ---------------------------------------------------------------------------
// Autocorrelation spectrum: one workgroup per (b,h).
// z = q + i*k packed complex FFT; accumulate Qf*conj(Kf) over dk; one IFFT.
// ---------------------------------------------------------------------------
__device__ void fft4096(float* zr, float* zi, float sign) {
  const int tid = threadIdx.x;
  for (int s = 1; s <= 12; ++s) {
    const int m = 1 << s, half = m >> 1;
    for (int j = tid; j < 2048; j += 256) {
      const int pos = j & (half - 1);
      const int i0  = ((j >> (s - 1)) << s) + pos;
      const int i1  = i0 + half;
      float ang = sign * 6.2831853071795864769f * (float)pos / (float)m;
      float wr, wi;
      __sincosf(ang, &wi, &wr);
      float xr = zr[i1], xi = zi[i1];
      float tr = wr * xr - wi * xi;
      float ti = wr * xi + wi * xr;
      float ur = zr[i0], ui = zi[i0];
      zr[i0] = ur + tr; zi[i0] = ui + ti;
      zr[i1] = ur - tr; zi[i1] = ui - ti;
    }
    __syncthreads();
  }
}

__global__ __launch_bounds__(256) void fft_corr_kernel(
    const float* __restrict__ Qt, const float* __restrict__ Kt,
    float* __restrict__ meanCorr) {
  __shared__ float zr[4096], zi[4096], ar[4096], ai[4096];  // 64 KB LDS
  const int ch  = blockIdx.x;  // b*H + h
  const int tid = threadIdx.x;
  const long long base = (long long)ch * DKn * Sn;
  for (int t = tid; t < 4096; t += 256) { ar[t] = 0.f; ai[t] = 0.f; }
  __syncthreads();
  for (int dk = 0; dk < DKn; ++dk) {
    const float* q = Qt + base + (long long)dk * Sn;
    const float* k = Kt + base + (long long)dk * Sn;
    for (int t = tid; t < 4096; t += 256) {        // bit-reversed load
      int r = __brev((unsigned)t) >> 20;
      zr[r] = q[t]; zi[r] = k[t];
    }
    __syncthreads();
    fft4096(zr, zi, -1.0f);                        // forward FFT
    for (int f = tid; f < 4096; f += 256) {        // cross-spectrum accumulate
      int g = (4096 - f) & 4095;
      float Zr = zr[f], Zi = zi[f], Wr = zr[g], Wi = zi[g];
      float qr = 0.5f * (Zr + Wr), qi = 0.5f * (Zi - Wi);   // Qf
      float kr = 0.5f * (Zi + Wi), ki = 0.5f * (Wr - Zr);   // Kf
      ar[f] += qr * kr + qi * ki;                  // Qf * conj(Kf)
      ai[f] += qi * kr - qr * ki;
    }
    __syncthreads();
  }
  for (int f = tid; f < 4096; f += 256) {          // IFFT of accumulated spec
    int r = __brev((unsigned)f) >> 20;
    zr[r] = ar[f]; zi[r] = ai[f];
  }
  __syncthreads();
  fft4096(zr, zi, 1.0f);
  const float scale = 1.0f / (4096.0f * (float)DKn);   // irfft 1/L and mean 1/DK
  for (int t = tid; t < 4096; t += 256)
    meanCorr[(long long)ch * Sn + t] = zr[t] * scale;
}

// ---------------------------------------------------------------------------
// top-8 + softmax per (b,h); writes attn to d_out tail
// ---------------------------------------------------------------------------
__global__ __launch_bounds__(256) void topk_softmax_kernel(
    const float* __restrict__ mean, int* __restrict__ delays,
    float* __restrict__ wgts, float* __restrict__ attnOut) {
  __shared__ float vals[4096];
  __shared__ float bv[256];
  __shared__ int   bi[256];
  __shared__ float topw[TOPK];
  __shared__ int   topi[TOPK];
  const int ch = blockIdx.x, tid = threadIdx.x;
  for (int t = tid; t < 4096; t += 256) vals[t] = mean[(long long)ch * Sn + t];
  __syncthreads();
  for (int i = 0; i < TOPK; ++i) {
    float best = -__builtin_inff(); int bidx = 1 << 30;
    for (int t = tid; t < 4096; t += 256) {
      float v = vals[t];
      if (v > best || (v == best && t < bidx)) { best = v; bidx = t; }
    }
    bv[tid] = best; bi[tid] = bidx;
    __syncthreads();
    for (int o = 128; o > 0; o >>= 1) {
      if (tid < o) {
        if (bv[tid + o] > bv[tid] ||
            (bv[tid + o] == bv[tid] && bi[tid + o] < bi[tid])) {
          bv[tid] = bv[tid + o]; bi[tid] = bi[tid + o];
        }
      }
      __syncthreads();
    }
    if (tid == 0) { topw[i] = bv[0]; topi[i] = bi[0]; vals[bi[0]] = -__builtin_inff(); }
    __syncthreads();
  }
  if (tid == 0) {
    float m = topw[0];
    for (int i = 1; i < TOPK; ++i) m = fmaxf(m, topw[i]);
    float e[TOPK], ssum = 0.f;
    for (int i = 0; i < TOPK; ++i) { e[i] = expf(topw[i] - m); ssum += e[i]; }
    for (int i = 0; i < TOPK; ++i) {
      float w = e[i] / ssum;
      wgts[ch * TOPK + i]    = w;
      attnOut[ch * TOPK + i] = w;
      delays[ch * TOPK + i]  = topi[i];
    }
  }
}

// ---------------------------------------------------------------------------
// agg[b,s,d] = sum_i w_i * v[b,(s+delay_i) mod S, d]   (bf16 out for Wo GEMM)
// ---------------------------------------------------------------------------
__global__ __launch_bounds__(256) void aggregate_kernel(
    const float* __restrict__ V, const int* __restrict__ delays,
    const float* __restrict__ wgts, bfraw* __restrict__ aggbf) {
  long long gid = (long long)blockIdx.x * 256 + threadIdx.x;
  if (gid >= BSD) return;
  int d = (int)(gid % Dn);
  long long bs = gid / Dn;
  int s = (int)(bs % Sn);
  int b = (int)(bs / Sn);
  int ch = b * Hn + (d / DKn);
  float acc = 0.f;
  #pragma unroll
  for (int i = 0; i < TOPK; ++i) {
    int   del = delays[ch * TOPK + i];
    float w   = wgts[ch * TOPK + i];
    int   ss  = (s + del) & (Sn - 1);
    acc += w * V[((long long)b * Sn + ss) * Dn + d];
  }
  aggbf[gid] = f2bf(acc);
}

// ---------------------------------------------------------------------------
// series decomposition: out = in - movavg25(in); optional bf16 copy (FFN in)
// ---------------------------------------------------------------------------
template<bool WRITE_BF>
__global__ __launch_bounds__(256) void decomp_kernel(
    const float* __restrict__ in, float* __restrict__ outF,
    bfraw* __restrict__ outB) {
  long long gid = (long long)blockIdx.x * 256 + threadIdx.x;
  if (gid >= BSD) return;
  int d = (int)(gid % Dn);
  long long bs = gid / Dn;
  int s = (int)(bs % Sn);
  int b = (int)(bs / Sn);
  const float* row = in + (long long)b * Sn * Dn + d;
  float sum = 0.f;
  #pragma unroll
  for (int j = -12; j <= 12; ++j) {
    int ss = s + j;
    ss = ss < 0 ? 0 : (ss > Sn - 1 ? Sn - 1 : ss);   // edge-replicate
    sum += row[(long long)ss * Dn];
  }
  float val = in[gid] - sum * (1.0f / 25.0f);
  outF[gid] = val;
  if (WRITE_BF) outB[gid] = f2bf(val);
}

// ---------------------------------------------------------------------------
// Host orchestration
// ---------------------------------------------------------------------------
extern "C" void kernel_launch(void* const* d_in, const int* in_sizes, int n_in,
                              void* d_out, int out_size, void* d_ws,
                              size_t ws_size, hipStream_t stream) {
  (void)in_sizes; (void)n_in; (void)out_size; (void)ws_size;
  const float* x  = (const float*)d_in[0];
  const float* Wq = (const float*)d_in[1];  const float* bq = (const float*)d_in[2];
  const float* Wk = (const float*)d_in[3];  const float* bk = (const float*)d_in[4];
  const float* Wv = (const float*)d_in[5];  const float* bv = (const float*)d_in[6];
  const float* Wo = (const float*)d_in[7];  const float* bo = (const float*)d_in[8];
  const float* W1 = (const float*)d_in[9];  const float* b1 = (const float*)d_in[10];
  const float* W2 = (const float*)d_in[11]; const float* b2 = (const float*)d_in[12];
  float* out = (float*)d_out;

  char* ws = (char*)d_ws;
  size_t off = 0;
  auto take = [&](size_t bytes) {
    size_t o = off; off += (bytes + 255) & ~(size_t)255; return o;
  };
  const size_t oWqb = take((size_t)512 * 512 * 2);
  const size_t oWkb = take((size_t)512 * 512 * 2);
  const size_t oWvb = take((size_t)512 * 512 * 2);
  const size_t oWob = take((size_t)512 * 512 * 2);
  const size_t oW1b = take((size_t)512 * 2048 * 2);
  const size_t oW2b = take((size_t)2048 * 512 * 2);
  const size_t oMean = take((size_t)64 * 4096 * 4);
  const size_t oDel  = take((size_t)64 * TOPK * 4);
  const size_t oWgt  = take((size_t)64 * TOPK * 4);
  const size_t oXbf  = take((size_t)BSD * 2);  // later reused as agg_bf
  const size_t oQd   = take((size_t)BSD * 4);  // later reused as y
  const size_t oKd   = take((size_t)BSD * 4);  // later reused as x1
  const size_t oVd   = take((size_t)BSD * 4);  // later reused as f
  const size_t oQt   = take((size_t)BSD * 4);  // later reused as x1_bf
  const size_t oKt   = take((size_t)BSD * 4);
  const size_t oH1   = take((size_t)Mrows * 2048 * 2);

  bfraw* Wqb  = (bfraw*)(ws + oWqb);
  bfraw* Wkb  = (bfraw*)(ws + oWkb);
  bfraw* Wvb  = (bfraw*)(ws + oWvb);
  bfraw* Wob  = (bfraw*)(ws + oWob);
  bfraw* W1b  = (bfraw*)(ws + oW1b);
  bfraw* W2b  = (bfraw*)(ws + oW2b);
  float* Mean = (float*)(ws + oMean);
  int*   Del  = (int*)(ws + oDel);
  float* Wgt  = (float*)(ws + oWgt);
  bfraw* Xbf  = (bfraw*)(ws + oXbf);
  float* Qd   = (float*)(ws + oQd);
  float* Kd   = (float*)(ws + oKd);
  float* Vd   = (float*)(ws + oVd);
  float* Qt   = (float*)(ws + oQt);
  float* Kt   = (float*)(ws + oKt);
  bfraw* AggBf = (bfraw*)(ws + oXbf);  // reuse (x_bf dead after QKV GEMMs)
  float* Y     = (float*)(ws + oQd);   // reuse (Qd dead after transpose)
  float* X1    = (float*)(ws + oKd);   // reuse (Kd dead after transpose)
  bfraw* X1bf  = (bfraw*)(ws + oQt);   // reuse (Qt dead after FFT)
  float* Fbuf  = (float*)(ws + oVd);   // reuse (Vd dead after aggregate)
  bfraw* H1bf  = (bfraw*)(ws + oH1);

  const dim3 blk(256);
  const int nBSD = (int)(BSD / 256);   // 65536

  // 1) bf16 casts
  cast_f32_bf16<<<nBSD, blk, 0, stream>>>(x, Xbf, BSD);
  cast_f32_bf16<<<(512 * 512 + 255) / 256, blk, 0, stream>>>(Wq, Wqb, 512 * 512);
  cast_f32_bf16<<<(512 * 512 + 255) / 256, blk, 0, stream>>>(Wk, Wkb, 512 * 512);
  cast_f32_bf16<<<(512 * 512 + 255) / 256, blk, 0, stream>>>(Wv, Wvb, 512 * 512);
  cast_f32_bf16<<<(512 * 512 + 255) / 256, blk, 0, stream>>>(Wo, Wob, 512 * 512);
  cast_f32_bf16<<<(512 * 2048 + 255) / 256, blk, 0, stream>>>(W1, W1b, 512 * 2048);
  cast_f32_bf16<<<(2048 * 512 + 255) / 256, blk, 0, stream>>>(W2, W2b, 2048 * 512);

  // 2) Q/K/V projections (bf16 WMMA, fp32 out)
  gemm_bf16_wmma<false, false, true, false>
      <<<dim3(4, 256), blk, 0, stream>>>(Xbf, Wqb, bq, nullptr, Qd, nullptr, 512, 512);
  gemm_bf16_wmma<false, false, true, false>
      <<<dim3(4, 256), blk, 0, stream>>>(Xbf, Wkb, bk, nullptr, Kd, nullptr, 512, 512);
  gemm_bf16_wmma<false, false, true, false>
      <<<dim3(4, 256), blk, 0, stream>>>(Xbf, Wvb, bv, nullptr, Vd, nullptr, 512, 512);

  // 3) transpose Q,K to (b,h,dk,s) for row-contiguous FFTs
  transpose_sd_kernel<<<dim3(128, 16, 8), dim3(32, 8), 0, stream>>>(Qd, Qt);
  transpose_sd_kernel<<<dim3(128, 16, 8), dim3(32, 8), 0, stream>>>(Kd, Kt);

  // 4) FFT autocorrelation -> mean over dk
  fft_corr_kernel<<<64, blk, 0, stream>>>(Qt, Kt, Mean);

  // 5) top-8 + softmax; attn written to d_out tail
  topk_softmax_kernel<<<64, blk, 0, stream>>>(Mean, Del, Wgt, out + BSD);

  // 6) time-delay aggregation (bf16 out)
  aggregate_kernel<<<nBSD, blk, 0, stream>>>(Vd, Del, Wgt, AggBf);

  // 7) y = agg @ Wo + bo + x
  gemm_bf16_wmma<false, true, true, false>
      <<<dim3(4, 256), blk, 0, stream>>>(AggBf, Wob, bo, x, Y, nullptr, 512, 512);

  // 8) x1 = y - movavg(y)  (+ bf16 copy for FFN)
  decomp_kernel<true><<<nBSD, blk, 0, stream>>>(Y, X1, X1bf);

  // 9) h1 = relu(x1 @ W1 + b1)  (bf16 out)
  gemm_bf16_wmma<true, false, false, true>
      <<<dim3(16, 256), blk, 0, stream>>>(X1bf, W1b, b1, nullptr, nullptr, H1bf, 2048, 512);

  // 10) f = h1 @ W2 + b2 + x1
  gemm_bf16_wmma<false, true, true, false>
      <<<dim3(4, 256), blk, 0, stream>>>(H1bf, W2b, b2, X1, Fbuf, nullptr, 512, 2048);

  // 11) res = f - movavg(f) -> d_out
  decomp_kernel<false><<<nBSD, blk, 0, stream>>>(Fbuf, out, nullptr);
}
